// Temporal_Attention_layer_53403623358619
// MI455X (gfx1250) — compile-verified, matrix operand order checked by reading, not run on hardware
//
#include <hip/hip_runtime.h>
#include <hip/hip_bf16.h>

// Problem constants (match reference)
#define Bc  4
#define Tc  2048
#define Dc  1024
#define Hc  16
#define DKc 64
#define DOc 1024

// GEMM pipeline config
#define KSTEP  64            // k-slice per stage (2 WMMA k-steps)
#define NSTAGE (Dc / KSTEP)  // 16
#define NBUF   4             // async pipeline depth (issue st+2, skew-safe)

typedef __attribute__((ext_vector_type(8)))  float  v8f;
typedef __attribute__((ext_vector_type(8)))  __bf16 v8bf;
typedef __attribute__((ext_vector_type(16))) __bf16 v16bf;

static __device__ __forceinline__ v16bf cat16(v8bf lo, v8bf hi) {
  return __builtin_shufflevector(lo, hi, 0,1,2,3,4,5,6,7,8,9,10,11,12,13,14,15);
}

#if defined(__HIP_DEVICE_COMPILE__) && \
    __has_builtin(__builtin_amdgcn_global_load_async_to_lds_b128) && \
    __has_builtin(__builtin_amdgcn_s_wait_asynccnt)
#define HAVE_ASYNC_LDS 1
#define GLOBAL_AS __attribute__((address_space(1)))
#define LDS_AS    __attribute__((address_space(3)))
#define WAIT_ASYNC(n) __builtin_amdgcn_s_wait_asynccnt(n)
typedef int v4i_g __attribute__((vector_size(16)));   // matches builtin pointee
static __device__ __forceinline__ void async_b128(const __bf16* g, __bf16* l) {
  __builtin_amdgcn_global_load_async_to_lds_b128(
      (GLOBAL_AS v4i_g*)(g), (LDS_AS v4i_g*)(l), 0, 0);
}
#else
#define HAVE_ASYNC_LDS 0
#endif

// ---------------------------------------------------------------------------
// Kernel 0: fp32 -> bf16 materialization (one-time streaming pass).
// ---------------------------------------------------------------------------
__global__ __launch_bounds__(256) void f32_to_bf16_kernel(
    const float* __restrict__ src, __bf16* __restrict__ dst, int n4)
{
  const int i = blockIdx.x * 256 + threadIdx.x;
  if (i < n4) {
    const float4 v = ((const float4*)src)[i];
    __bf16* d = dst + (size_t)i * 4;
    d[0] = (__bf16)v.x; d[1] = (__bf16)v.y;
    d[2] = (__bf16)v.z; d[3] = (__bf16)v.w;
  }
}

// ---------------------------------------------------------------------------
// Kernel 1: K/V projection from pre-converted bf16 operands.
// Y[m,n] = sum_d x[m,d] * W[n,d] + bias[n];  M=8192, N=1024, K=1024.
// WG tile 128x64, 8 waves of 32x32 (2x2 WMMA tiles), K-step 64.
// Async GLOBAL->LDS staging, 4-deep pipeline, s_wait_asynccnt throttling.
// ---------------------------------------------------------------------------
__global__ __launch_bounds__(256) void proj_kv_kernel(
    const __bf16* __restrict__ xg,                                   // 8192x1024
    const __bf16* __restrict__ Wkb, const __bf16* __restrict__ Wvb,  // 1024x1024
    const float* __restrict__ bk, const float* __restrict__ bv,
    __bf16* __restrict__ Kout, __bf16* __restrict__ Vout)
{
  const __bf16* W    = (blockIdx.z == 0) ? Wkb : Wvb;
  const float*  bias = (blockIdx.z == 0) ? bk : bv;
  __bf16*       Y    = (blockIdx.z == 0) ? Kout : Vout;

  const int Mbase = blockIdx.y * 128;
  const int Nbase = blockIdx.x * 64;

  __shared__ __align__(16) __bf16 xs[NBUF][128 * KSTEP];  // 16 KB / buf
  __shared__ __align__(16) __bf16 wl[NBUF][64 * KSTEP];   //  8 KB / buf

  const int tid  = threadIdx.x;
  const int wave = tid >> 5;
  const int lane = tid & 31;
  const int half = lane >> 4;   // 0 or 1
  const int idx  = lane & 15;
  const int wm   = wave & 3;    // 4 M-slots * 32 = 128
  const int wn   = wave >> 2;   // 2 N-slots * 32 = 64

  v8f acc[2][2] = {};

  auto compute = [&](int bufi) {
    const __bf16* xb = xs[bufi];
    const __bf16* wb = wl[bufi];
    #pragma unroll
    for (int ks = 0; ks < 2; ++ks) {           // two k=32 steps per stage
      v16bf a[2], b[2];
      #pragma unroll
      for (int mi = 0; mi < 2; ++mi) {
        const int r = wm * 32 + mi * 16 + idx;
        const v8bf lo = *(const v8bf*)&xb[r * KSTEP + ks * 32 + half * 8];
        const v8bf hi = *(const v8bf*)&xb[r * KSTEP + ks * 32 + 16 + half * 8];
        a[mi] = cat16(lo, hi);
      }
      #pragma unroll
      for (int ni = 0; ni < 2; ++ni) {
        const int r = wn * 32 + ni * 16 + idx;
        const v8bf lo = *(const v8bf*)&wb[r * KSTEP + ks * 32 + half * 8];
        const v8bf hi = *(const v8bf*)&wb[r * KSTEP + ks * 32 + 16 + half * 8];
        b[ni] = cat16(lo, hi);
      }
      #pragma unroll
      for (int mi = 0; mi < 2; ++mi)
        #pragma unroll
        for (int ni = 0; ni < 2; ++ni)
          acc[mi][ni] = __builtin_amdgcn_wmma_f32_16x16x32_bf16(
              false, a[mi], false, b[ni], (short)0, acc[mi][ni], false, false);
    }
  };

#if HAVE_ASYNC_LDS
  auto issue = [&](int st) {                   // 6 async b128 per thread
    const int kk   = st * KSTEP;
    const int bufi = st & (NBUF - 1);
    #pragma unroll
    for (int i = 0; i < 4; ++i) {
      const int t = tid + i * 256;
      const int row = t >> 3;
      const int c8  = (t & 7) << 3;
      async_b128(xg + (size_t)(Mbase + row) * Dc + kk + c8,
                 &xs[bufi][row * KSTEP + c8]);
    }
    #pragma unroll
    for (int i = 0; i < 2; ++i) {
      const int t = tid + i * 256;
      const int row = t >> 3;
      const int c8  = (t & 7) << 3;
      async_b128(W + (size_t)(Nbase + row) * Dc + kk + c8,
                 &wl[bufi][row * KSTEP + c8]);
    }
  };

  issue(0);
  issue(1);
  for (int st = 0; st < NSTAGE; ++st) {
    if (st + 2 < NSTAGE) { issue(st + 2); WAIT_ASYNC(12); }  // 2 stages in flight
    else if (st + 1 < NSTAGE) { WAIT_ASYNC(6); }             // 1 stage in flight
    else { WAIT_ASYNC(0); }
    __syncthreads();
    compute(st & (NBUF - 1));
  }
#else
  // Synchronous fallback: bf16 global -> VGPR -> LDS, single buffer.
  for (int st = 0; st < NSTAGE; ++st) {
    const int kk = st * KSTEP;
    __syncthreads();
    #pragma unroll
    for (int i = 0; i < 4; ++i) {
      const int t = tid + i * 256;
      const int row = t >> 3;
      const int c8  = (t & 7) << 3;
      *(v8bf*)&xs[0][row * KSTEP + c8] =
          *(const v8bf*)(xg + (size_t)(Mbase + row) * Dc + kk + c8);
    }
    #pragma unroll
    for (int i = 0; i < 2; ++i) {
      const int t = tid + i * 256;
      const int row = t >> 3;
      const int c8  = (t & 7) << 3;
      *(v8bf*)&wl[0][row * KSTEP + c8] =
          *(const v8bf*)(W + (size_t)(Nbase + row) * Dc + kk + c8);
    }
    __syncthreads();
    compute(0);
  }
#endif

  // Epilogue: C layout VGPR r -> M = r + 8*half, N = idx.
  #pragma unroll
  for (int mi = 0; mi < 2; ++mi) {
    #pragma unroll
    for (int ni = 0; ni < 2; ++ni) {
      const int col  = Nbase + wn * 32 + ni * 16 + idx;
      const float bb = bias[col];
      const int rowbase = Mbase + wm * 32 + mi * 16 + half * 8;
      #pragma unroll
      for (int r = 0; r < 8; ++r)
        Y[(size_t)(rowbase + r) * DOc + col] = (__bf16)(acc[mi][ni][r] + bb);
    }
  }
}

// ---------------------------------------------------------------------------
// Kernel 2: tiny GEMM, <=8 rows (M padded to 16 with zeros), one wave/N-tile.
// out[r,n] = sum_k A[r,k] * W[n,k] + bias[n].  Used for q-last and out-proj.
// ---------------------------------------------------------------------------
__global__ __launch_bounds__(32) void gemm_rows16_kernel(
    const float* __restrict__ A, size_t arow_stride, int arows,
    const float* __restrict__ W, const float* __restrict__ bias,
    float* __restrict__ out, size_t orow_stride)
{
  const int Nbase = blockIdx.x * 16;
  const int lane  = threadIdx.x & 31;
  const int half  = lane >> 4;
  const int idx   = lane & 15;

  v8f acc = {};
  for (int kk = 0; kk < Dc; kk += 32) {
    v16bf a, b;
    const int k0 = kk + half * 8;
    #pragma unroll
    for (int j = 0; j < 8; ++j) {
      const float a0 = (idx < arows) ? A[(size_t)idx * arow_stride + k0 + j]      : 0.0f;
      const float a1 = (idx < arows) ? A[(size_t)idx * arow_stride + k0 + 16 + j] : 0.0f;
      a[j]     = (__bf16)a0;
      a[8 + j] = (__bf16)a1;
      b[j]     = (__bf16)W[(size_t)(Nbase + idx) * Dc + k0 + j];
      b[8 + j] = (__bf16)W[(size_t)(Nbase + idx) * Dc + k0 + 16 + j];
    }
    acc = __builtin_amdgcn_wmma_f32_16x16x32_bf16(
        false, a, false, b, (short)0, acc, false, false);
  }

  if (half == 0) {
    const float bb = bias[Nbase + idx];
    #pragma unroll
    for (int r = 0; r < 8; ++r)
      if (r < arows)
        out[(size_t)r * orow_stride + Nbase + idx] = acc[r] + bb;
  }
}

// ---------------------------------------------------------------------------
// Kernel 3: last-token attention, one block per (b, h).
// scores[s] = (q . K[s]) / 8 ; softmax over T ; ctx = sum_s attn[s] * V[s].
// ---------------------------------------------------------------------------
__global__ __launch_bounds__(256) void attn_last_kernel(
    const float* __restrict__ qws, const __bf16* __restrict__ Kws,
    const __bf16* __restrict__ Vws, float* __restrict__ ctxws)
{
  const int b   = blockIdx.x >> 4;   // / H
  const int h   = blockIdx.x & 15;   // % H
  const int tid = threadIdx.x;

  __shared__ float qs[DKc];
  __shared__ float wls[Tc];          // 8 KB attn weights
  __shared__ float red[256];
  __shared__ float cpart[4][DKc];

  if (tid < DKc) qs[tid] = qws[b * DOc + h * DKc + tid];
  __syncthreads();

  float sc[8];
  float mymax = -1e30f;
  #pragma unroll
  for (int i = 0; i < 8; ++i) {
    const int s = tid + i * 256;
    const v8bf* kp = (const v8bf*)(Kws + ((size_t)(b * Tc + s)) * DOc + h * DKc);
    float d0 = 0.f;
    #pragma unroll
    for (int c = 0; c < 8; ++c) {
      const v8bf kv = kp[c];
      #pragma unroll
      for (int j = 0; j < 8; ++j) d0 += qs[c * 8 + j] * (float)kv[j];
    }
    d0 *= 0.125f;                    // 1/sqrt(DK)
    sc[i] = d0;
    mymax = fmaxf(mymax, d0);
  }

  red[tid] = mymax; __syncthreads();
  for (int off = 128; off > 0; off >>= 1) {
    if (tid < off) red[tid] = fmaxf(red[tid], red[tid + off]);
    __syncthreads();
  }
  const float m = red[0];
  __syncthreads();

  float mysum = 0.f;
  #pragma unroll
  for (int i = 0; i < 8; ++i) {
    const float e = __expf(sc[i] - m);
    wls[tid + i * 256] = e;
    mysum += e;
  }
  red[tid] = mysum; __syncthreads();
  for (int off = 128; off > 0; off >>= 1) {
    if (tid < off) red[tid] = red[tid] + red[tid + off];
    __syncthreads();
  }
  const float inv = 1.0f / red[0];

  const int d     = tid & 63;
  const int chunk = tid >> 6;        // 0..3, each covers 512 of T
  float a = 0.f;
  for (int s = chunk * 512; s < (chunk + 1) * 512; ++s)
    a += wls[s] * (float)Vws[((size_t)(b * Tc + s)) * DOc + h * DKc + d];
  cpart[chunk][d] = a;
  __syncthreads();

  if (tid < DKc) {
    const float r = (cpart[0][tid] + cpart[1][tid] + cpart[2][tid] + cpart[3][tid]) * inv;
    ctxws[b * DOc + h * DKc + tid] = r;
  }
}

// ---------------------------------------------------------------------------
extern "C" void kernel_launch(void* const* d_in, const int* in_sizes, int n_in,
                              void* d_out, int out_size, void* d_ws, size_t ws_size,
                              hipStream_t stream) {
  (void)in_sizes; (void)n_in; (void)out_size; (void)ws_size;

  const float* x  = (const float*)d_in[0];
  const float* Wq = (const float*)d_in[1];
  const float* bq = (const float*)d_in[2];
  const float* Wk = (const float*)d_in[3];
  const float* bk = (const float*)d_in[4];
  const float* Wv = (const float*)d_in[5];
  const float* bv = (const float*)d_in[6];
  const float* Wo = (const float*)d_in[7];
  const float* bo = (const float*)d_in[8];
  float* out = (float*)d_out;

  char* ws = (char*)d_ws;
  const size_t kvBytes = (size_t)Bc * Tc * DOc * sizeof(__bf16);  // 16 MB
  const size_t wBytes  = (size_t)DOc * Dc * sizeof(__bf16);       //  2 MB
  __bf16* Kb    = (__bf16*)(ws);
  __bf16* Vb    = (__bf16*)(ws + kvBytes);
  __bf16* Xb    = (__bf16*)(ws + 2 * kvBytes);                    // 16 MB
  __bf16* Wkb   = (__bf16*)(ws + 3 * kvBytes);
  __bf16* Wvb   = (__bf16*)(ws + 3 * kvBytes + wBytes);
  float*  qws   = (float*)(ws + 3 * kvBytes + 2 * wBytes);
  float*  ctxws = qws + Bc * DOc;

  // 0) one-time bf16 materialization of x, Wk, Wv
  const int xq4 = (Bc * Tc * Dc) / 4;   // 2,097,152
  const int wq4 = (DOc * Dc) / 4;       //   262,144
  f32_to_bf16_kernel<<<(xq4 + 255) / 256, 256, 0, stream>>>(x,  Xb,  xq4);
  f32_to_bf16_kernel<<<(wq4 + 255) / 256, 256, 0, stream>>>(Wk, Wkb, wq4);
  f32_to_bf16_kernel<<<(wq4 + 255) / 256, 256, 0, stream>>>(Wv, Wvb, wq4);

  // 1) K and V projections (bulk of the FLOPs) -> bf16 workspace
  dim3 gp(DOc / 64, (Bc * Tc) / 128, 2);
  proj_kv_kernel<<<gp, 256, 0, stream>>>(Xb, Wkb, Wvb, bk, bv, Kb, Vb);

  // 2) Q for the last token of each batch (4 rows)
  gemm_rows16_kernel<<<DOc / 16, 32, 0, stream>>>(
      x + (size_t)(Tc - 1) * Dc, (size_t)Tc * Dc, Bc, Wq, bq, qws, DOc);

  // 3) softmax attention for the last token, one block per (b, h)
  attn_last_kernel<<<Bc * Hc, 256, 0, stream>>>(qws, Kb, Vb, ctxws);

  // 4) output projection (4 rows) -> d_out
  gemm_rows16_kernel<<<DOc / 16, 32, 0, stream>>>(
      ctxws, DOc, Bc, Wo, bo, out, DOc);
}
